// SocialAttention_16956530885226
// MI455X (gfx1250) — matrix operand order repaired and checked by reading, compiled, not run on hardware
//
#include <hip/hip_runtime.h>
#include <hip/hip_bf16.h>

typedef __attribute__((ext_vector_type(16))) _Float16 v16h;
typedef __attribute__((ext_vector_type(8)))  _Float16 v8h;
typedef __attribute__((ext_vector_type(4)))  _Float16 v4h;
typedef __attribute__((ext_vector_type(8)))  float    v8f;
typedef __attribute__((ext_vector_type(4)))  float    v4f;

#define B_   2
#define S_   2048
#define HID_ 1024
#define H_   16
#define D_   64
#define MROWS 4096   // B*S

// ---------- WMMA fragment helpers (layouts per CDNA5 ISA 7.12.2) ----------

__device__ __forceinline__ v16h cat16(v8h lo, v8h hi) {
  return __builtin_shufflevector(lo, hi, 0,1,2,3,4,5,6,7,8,9,10,11,12,13,14,15);
}

// A fragment 16x32 f16: lane = row (lane&15); halves j:0..7 -> k = hi8 + j,
// j:8..15 -> k = 16 + hi8 + (j-8), hi8 = (lane>=16)?8:0.  Source row-major
// [m][k] with row stride `ld` halves (ld*2 bytes must be a 16B multiple).
__device__ __forceinline__ v16h frag_a(const _Float16* tile, int ld) {
  const int lane = threadIdx.x & 31;
  const _Float16* p = tile + (lane & 15) * ld + ((lane >> 4) << 3);
  v8h lo = *(const v8h*)(p);
  v8h hi = *(const v8h*)(p + 16);
  return cat16(lo, hi);
}

// B fragment 32x16 f16: lane = col (lane&15); halves j:0..7 -> k = hi8 + j, j:8..15 -> k = 16 + hi8 + (j-8).
// Source stored as [n][k] row-major (i.e. B transposed) with row stride `ld`.
__device__ __forceinline__ v16h frag_b(const _Float16* tile, int ld) {
  const int lane = threadIdx.x & 31;
  const _Float16* p = tile + (lane & 15) * ld + ((lane >> 4) << 3);
  v8h lo = *(const v8h*)(p);
  v8h hi = *(const v8h*)(p + 16);
  return cat16(lo, hi);
}

__device__ __forceinline__ v8f wmma_f16(v16h a, v16h b, v8f c) {
  return __builtin_amdgcn_wmma_f32_16x16x32_f16(false, a, false, b, (short)0, c,
                                                false, false);
}

__device__ __forceinline__ v4h cvt4(v4f f) {
  v4h h;
  #pragma unroll
  for (int j = 0; j < 4; ++j) h[j] = (_Float16)f[j];
  return h;
}

// ---------- Kernel 1: fused QKV projection GEMM (z picks Q/K/V) ----------
// X [4096,1024] f32 @ W [1024,1024] f32 + b -> head-split f16 [B,H,S,D]
// BK=64: 8 WMMAs per barrier pair; all staging via 16B global loads.
__global__ __launch_bounds__(256) void qkv_gemm_kernel(
    const float* __restrict__ X,
    const float* __restrict__ Wq, const float* __restrict__ bq,
    const float* __restrict__ Wk, const float* __restrict__ bk,
    const float* __restrict__ Wv, const float* __restrict__ bv,
    _Float16* __restrict__ Qw, _Float16* __restrict__ Kw,
    _Float16* __restrict__ Vw) {
  const int z = blockIdx.z;
  const float* W    = (z == 0) ? Wq : (z == 1) ? Wk : Wv;
  const float* bias = (z == 0) ? bq : (z == 1) ? bk : bv;
  _Float16*    dst  = (z == 0) ? Qw : (z == 1) ? Kw : Vw;

  const int n0 = blockIdx.x * 64;
  const int m0 = blockIdx.y * 128;
  const int tid = threadIdx.x;
  const int wave = tid >> 5, lane = tid & 31;
  const int wm = wave >> 1, wn = wave & 1;   // 4x2 wave grid, 32x32 per wave

  __shared__ __align__(16) _Float16 a_lds[128 * 72];  // 128 x 64, stride 72
  __shared__ __align__(16) _Float16 b_lds[64 * 72];   // [n=64][k=64], stride 72

  v8f acc[2][2] = {};

  for (int kk = 0; kk < HID_; kk += 64) {
    // A tile 128x64 f32 -> f16, float4 loads, v4h stores
    #pragma unroll
    for (int i = 0; i < 8; ++i) {
      int c = tid + i * 256;              // 2048 chunks of 4
      int r = c >> 4, c4 = (c & 15) * 4;
      v4f f = *(const v4f*)&X[(size_t)(m0 + r) * HID_ + kk + c4];
      *(v4h*)&a_lds[r * 72 + c4] = cvt4(f);
    }
    // B tile 64x64 -> LDS transposed [n][k], float4 global reads along n
    #pragma unroll
    for (int i = 0; i < 4; ++i) {
      int c = tid + i * 256;              // 1024 chunks of 4
      int n4 = (c & 15) * 4, k = c >> 4;
      v4f f = *(const v4f*)&W[(size_t)(kk + k) * HID_ + n0 + n4];
      #pragma unroll
      for (int j = 0; j < 4; ++j) b_lds[(n4 + j) * 72 + k] = (_Float16)f[j];
    }
    __syncthreads();
    if (kk + 64 < HID_) {
      __builtin_prefetch(&X[(size_t)(m0 + wave * 16) * HID_ + kk + 64], 0, 0);
      __builtin_prefetch(&W[(size_t)(kk + 64 + wave * 8) * HID_ + n0], 0, 0);
    }
    #pragma unroll
    for (int kc = 0; kc < 2; ++kc) {
      v16h a0 = frag_a(&a_lds[(wm * 32 + 0)  * 72 + kc * 32], 72);
      v16h a1 = frag_a(&a_lds[(wm * 32 + 16) * 72 + kc * 32], 72);
      v16h b0 = frag_b(&b_lds[(wn * 32 + 0)  * 72 + kc * 32], 72);
      v16h b1 = frag_b(&b_lds[(wn * 32 + 16) * 72 + kc * 32], 72);
      acc[0][0] = wmma_f16(a0, b0, acc[0][0]);
      acc[0][1] = wmma_f16(a0, b1, acc[0][1]);
      acc[1][0] = wmma_f16(a1, b0, acc[1][0]);
      acc[1][1] = wmma_f16(a1, b1, acc[1][1]);
    }
    __syncthreads();
  }

  const int mrow_hi = (lane >> 4) << 3;
  #pragma unroll
  for (int ti = 0; ti < 2; ++ti)
    #pragma unroll
    for (int tj = 0; tj < 2; ++tj)
      #pragma unroll
      for (int r = 0; r < 8; ++r) {
        int m = m0 + wm * 32 + ti * 16 + r + mrow_hi;   // row in [0,4096)
        int n = n0 + wn * 32 + tj * 16 + (lane & 15);   // col in [0,1024)
        float v = acc[ti][tj][r] + bias[n];
        int bb = m >> 11, ss = m & 2047, hh = n >> 6, dd = n & 63;
        dst[(((size_t)(bb * H_ + hh) * S_ + ss) << 6) + dd] = (_Float16)v;
      }
}

// ---------- Kernel 2: fused scores + softmax + probs-out + probs@V ----------
// One block: 32 queries x one (b,h).  Dynamic LDS:
//   sc : 32 x 2048 f16 score/prob slab            = 131072 B
//   vst: 64 x 136 f16 V-chunk (128 keys), [d][key] =  17408 B
__global__ __launch_bounds__(256) void attn_fused_kernel(
    const _Float16* __restrict__ Qw, const _Float16* __restrict__ Kw,
    const _Float16* __restrict__ Vw,
    const float* __restrict__ amask,
    const float* __restrict__ hm, const float* __restrict__ mmk,
    const float* __restrict__ emk,
    const float* __restrict__ hb, const float* __restrict__ mb,
    const float* __restrict__ eb,
    float* __restrict__ probs, _Float16* __restrict__ Cw) {
  extern __shared__ __align__(16) char smem[];
  _Float16* sc  = (_Float16*)smem;                    // [32][2048]
  _Float16* vst = (_Float16*)(smem + 32 * 2048 * 2);  // [64][136]

  const int tid = threadIdx.x;
  const int wave = tid >> 5, lane = tid & 31;
  const int bh = blockIdx.y;
  const int b = bh >> 4, h = bh & 15;
  const int q0 = blockIdx.x * 32;
  const int mrow_hi = (lane >> 4) << 3;

  const _Float16* qbase = Qw + ((size_t)bh * S_ + q0) * D_;
  const _Float16* kbase = Kw + (size_t)bh * S_ * D_;
  const _Float16* vbase = Vw + (size_t)bh * S_ * D_;

  // ---- Phase 1: scores = QK^T/8 + biases, f16 into LDS ----
  v16h aq[2][2];
  aq[0][0] = frag_a(qbase,                D_);
  aq[0][1] = frag_a(qbase + 32,           D_);
  aq[1][0] = frag_a(qbase + 16 * D_,      D_);
  aq[1][1] = frag_a(qbase + 16 * D_ + 32, D_);

  const float hbv = hb[h], mbv = mb[h], ebv = eb[h];
  for (int kt = wave; kt < S_ / 16; kt += 8) {
    const int key0 = kt * 16;
    v16h b0 = frag_b(kbase + (size_t)key0 * D_,      D_);
    v16h b1 = frag_b(kbase + (size_t)key0 * D_ + 32, D_);
    const int key = key0 + (lane & 15);
    const float t = hbv * hm[b * S_ + key] + mbv * mmk[b * S_ + key] +
                    ebv * emk[b * S_ + key] + amask[b * S_ + key];
    #pragma unroll
    for (int qt = 0; qt < 2; ++qt) {
      v8f c = {};
      c = wmma_f16(aq[qt][0], b0, c);
      c = wmma_f16(aq[qt][1], b1, c);
      #pragma unroll
      for (int r = 0; r < 8; ++r)
        sc[(size_t)(qt * 16 + r + mrow_hi) * S_ + key] =
            (_Float16)(c[r] * 0.125f + t);
    }
  }
  __syncthreads();

  // ---- Phase 2: per-row softmax in LDS; probs streamed as b128 stores ----
  for (int rr = 0; rr < 4; ++rr) {
    const int row = wave * 4 + rr;            // wave owns whole rows
    _Float16* prow = sc + (size_t)row * S_;
    float mx = -3.4e38f;
    #pragma unroll
    for (int i = 0; i < S_ / 128; ++i) {      // 16 x v4h loads
      v4h hv = *(const v4h*)&prow[lane * 4 + i * 128];
      #pragma unroll
      for (int j = 0; j < 4; ++j) mx = fmaxf(mx, (float)hv[j]);
    }
    #pragma unroll
    for (int off = 16; off > 0; off >>= 1) mx = fmaxf(mx, __shfl_xor(mx, off));
    float s = 0.f;
    #pragma unroll
    for (int i = 0; i < S_ / 128; ++i) {
      v4h hv = *(const v4h*)&prow[lane * 4 + i * 128];
      v4h ev;
      #pragma unroll
      for (int j = 0; j < 4; ++j) {
        float e = __expf((float)hv[j] - mx);
        s += e;
        ev[j] = (_Float16)e;
      }
      *(v4h*)&prow[lane * 4 + i * 128] = ev;
    }
    #pragma unroll
    for (int off = 16; off > 0; off >>= 1) s += __shfl_xor(s, off);
    const float inv = 1.0f / s;
    float* gp = probs + ((size_t)bh * S_ + q0 + row) * S_;
    #pragma unroll
    for (int i = 0; i < S_ / 128; ++i) {
      v4h ev = *(const v4h*)&prow[lane * 4 + i * 128];
      v4f pv;
      #pragma unroll
      for (int j = 0; j < 4; ++j) pv[j] = (float)ev[j] * inv;
      *(v4f*)&gp[lane * 4 + i * 128] = pv;    // probs output: coalesced b128
      *(v4h*)&prow[lane * 4 + i * 128] = cvt4(pv);  // normalized f16, phase 3
    }
  }
  __syncthreads();

  // ---- Phase 3: ctx = probs @ V, A-fragments straight from LDS slab ----
  const int qt = wave >> 2;                   // 0..1 : which 16 query rows
  const int nt = wave & 3;                    // 0..3 : which 16 of D=64 cols
  v8f acc = {};
  for (int kk = 0; kk < S_; kk += 128) {
    #pragma unroll
    for (int i = 0; i < 8; ++i) {             // V 128keys x 64d -> [d][key]
      int c = tid + i * 256;                  // 2048 chunks of 4
      int n4 = (c & 15) * 4, k = c >> 4;
      v4h hv = *(const v4h*)&vbase[(size_t)(kk + k) * D_ + n4];
      #pragma unroll
      for (int j = 0; j < 4; ++j) vst[(n4 + j) * 136 + k] = hv[j];
    }
    __syncthreads();
    if (kk + 128 < S_)
      __builtin_prefetch(&vbase[(size_t)(kk + 128 + wave * 16) * D_], 0, 0);
    #pragma unroll
    for (int kc = 0; kc < 4; ++kc) {
      v16h a = frag_a(sc + (size_t)(qt * 16) * S_ + kk + kc * 32, S_);
      v16h v = frag_b(vst + (nt * 16) * 136 + kc * 32, 136);
      acc = wmma_f16(a, v, acc);
    }
    __syncthreads();
  }
  #pragma unroll
  for (int r = 0; r < 8; ++r) {
    const int q = q0 + qt * 16 + r + mrow_hi;
    const int d = nt * 16 + (lane & 15);
    Cw[(size_t)(b * S_ + q) * HID_ + h * D_ + d] = (_Float16)acc[r];
  }
}

// ---------- Kernel 3: out = ctx @ Wo + bo (BK=64) ----------
__global__ __launch_bounds__(256) void oproj_gemm_kernel(
    const _Float16* __restrict__ Cw, const float* __restrict__ Wo,
    const float* __restrict__ bo, float* __restrict__ out) {
  const int n0 = blockIdx.x * 64;
  const int m0 = blockIdx.y * 128;
  const int tid = threadIdx.x, wave = tid >> 5, lane = tid & 31;
  const int wm = wave >> 1, wn = wave & 1;

  __shared__ __align__(16) _Float16 b_lds[64 * 72];   // [n=64][k=64]

  v8f acc[2][2] = {};
  for (int kk = 0; kk < HID_; kk += 64) {
    #pragma unroll
    for (int i = 0; i < 4; ++i) {
      int c = tid + i * 256;
      int n4 = (c & 15) * 4, k = c >> 4;
      v4f f = *(const v4f*)&Wo[(size_t)(kk + k) * HID_ + n0 + n4];
      #pragma unroll
      for (int j = 0; j < 4; ++j) b_lds[(n4 + j) * 72 + k] = (_Float16)f[j];
    }
    __syncthreads();
    if (kk + 64 < HID_)
      __builtin_prefetch(&Wo[(size_t)(kk + 64 + wave * 8) * HID_ + n0], 0, 0);
    #pragma unroll
    for (int kc = 0; kc < 2; ++kc) {
      // A fragments straight from global f16 ctx (row stride 1024 halves)
      v16h a0 = frag_a(Cw + (size_t)(m0 + wm * 32)      * HID_ + kk + kc * 32, HID_);
      v16h a1 = frag_a(Cw + (size_t)(m0 + wm * 32 + 16) * HID_ + kk + kc * 32, HID_);
      v16h b0 = frag_b(&b_lds[(wn * 32 + 0)  * 72 + kc * 32], 72);
      v16h b1 = frag_b(&b_lds[(wn * 32 + 16) * 72 + kc * 32], 72);
      acc[0][0] = wmma_f16(a0, b0, acc[0][0]);
      acc[0][1] = wmma_f16(a0, b1, acc[0][1]);
      acc[1][0] = wmma_f16(a1, b0, acc[1][0]);
      acc[1][1] = wmma_f16(a1, b1, acc[1][1]);
    }
    __syncthreads();
  }
  const int mrow_hi = (lane >> 4) << 3;
  #pragma unroll
  for (int ti = 0; ti < 2; ++ti)
    #pragma unroll
    for (int tj = 0; tj < 2; ++tj)
      #pragma unroll
      for (int r = 0; r < 8; ++r) {
        int m = m0 + wm * 32 + ti * 16 + r + mrow_hi;
        int n = n0 + wn * 32 + tj * 16 + (lane & 15);
        out[(size_t)m * HID_ + n] = acc[ti][tj][r] + bo[n];
      }
}

// ---------- host side ----------
extern "C" void kernel_launch(void* const* d_in, const int* in_sizes, int n_in,
                              void* d_out, int out_size, void* d_ws,
                              size_t ws_size, hipStream_t stream) {
  const float* hs   = (const float*)d_in[0];
  const float* amsk = (const float*)d_in[1];
  const float* hm   = (const float*)d_in[2];
  const float* mmk  = (const float*)d_in[3];
  const float* emk  = (const float*)d_in[4];
  const float* Wq   = (const float*)d_in[5];
  const float* bq   = (const float*)d_in[6];
  const float* Wk   = (const float*)d_in[7];
  const float* bk   = (const float*)d_in[8];
  const float* Wv   = (const float*)d_in[9];
  const float* bv   = (const float*)d_in[10];
  const float* Wo   = (const float*)d_in[11];
  const float* bo   = (const float*)d_in[12];
  const float* hb   = (const float*)d_in[13];
  const float* mb   = (const float*)d_in[14];
  const float* eb   = (const float*)d_in[15];

  float* out   = (float*)d_out;
  float* probs = out + (size_t)B_ * S_ * HID_;   // second tuple output

  // Workspace: Q,K,V head-split f16 + ctx f16  = 4 * 8MB = 32MB
  _Float16* Qw = (_Float16*)d_ws;
  _Float16* Kw = Qw + (size_t)MROWS * HID_;
  _Float16* Vw = Kw + (size_t)MROWS * HID_;
  _Float16* Cw = Vw + (size_t)MROWS * HID_;

  qkv_gemm_kernel<<<dim3(HID_ / 64, MROWS / 128, 3), 256, 0, stream>>>(
      hs, Wq, bq, Wk, bk, Wv, bv, Qw, Kw, Vw);

  const size_t attn_lds = (size_t)32 * 2048 * 2 + (size_t)64 * 136 * 2; // 148480B
  attn_fused_kernel<<<dim3(S_ / 32, B_ * H_), 256, attn_lds, stream>>>(
      Qw, Kw, Vw, amsk, hm, mmk, emk, hb, mb, eb, probs, Cw);

  oproj_gemm_kernel<<<dim3(HID_ / 64, MROWS / 128), 256, 0, stream>>>(
      Cw, Wo, bo, out);
}
